// VAE_GAT_77936476553830
// MI455X (gfx1250) — matrix operand order repaired and checked by reading, not compiled, not run on hardware
//
#include <hip/hip_runtime.h>
#include <hip/hip_bf16.h>
#include <cstdint>

// ---------------------------------------------------------------------------
// Types for CDNA5 WMMA (wave32): v_wmma_f32_16x16x32_bf16
// ---------------------------------------------------------------------------
typedef __attribute__((ext_vector_type(16))) __bf16 v16bf;
typedef __attribute__((ext_vector_type(8)))  float  v8f;

#define WAVE 32

__device__ __forceinline__ unsigned short f2bf(float f) {
    unsigned u = __float_as_uint(f);
    unsigned r = (u + 0x7FFFu + ((u >> 16) & 1u)) >> 16;   // round-to-nearest-even
    return (unsigned short)r;
}

// monotone map float -> uint so unsigned atomicMax == float max
__device__ __forceinline__ unsigned fmap(float x) {
    unsigned u = __float_as_uint(x);
    return (u & 0x80000000u) ? ~u : (u | 0x80000000u);
}
__device__ __forceinline__ float funmap(unsigned u) {
    return __uint_as_float((u & 0x80000000u) ? (u & 0x7FFFFFFFu) : ~u);
}

// ---------------------------------------------------------------------------
// Utility kernels
// ---------------------------------------------------------------------------
__global__ void zero_f32(float* p, size_t n) {
    for (size_t i = blockIdx.x * (size_t)blockDim.x + threadIdx.x; i < n;
         i += (size_t)gridDim.x * blockDim.x)
        p[i] = 0.0f;
}

__global__ void cvt_f32_bf16(const float* __restrict__ in, unsigned short* __restrict__ out,
                             size_t n) {
    for (size_t i = blockIdx.x * (size_t)blockDim.x + threadIdx.x; i < n;
         i += (size_t)gridDim.x * blockDim.x)
        out[i] = f2bf(in[i]);
}

// Pack W[NI,NO] fp32 (row-major) into WMMA B-operand lane layout, bf16.
// Flat index = ((t*KB + kb)*32 + lane)*16 + j
//   lane<16 : value = W[kb*32 +     (j<8 ? j : j+8)][t*16 + lane]
//   lane>=16: value = W[kb*32 + 8 + (j<8 ? j : j+8)][t*16 + lane-16]
__global__ void pack_w(const float* __restrict__ W, unsigned short* __restrict__ out,
                       int NI, int NO) {
    int total = NI * NO;
    int KB = NI / 32;
    for (int tid = blockIdx.x * blockDim.x + threadIdx.x; tid < total;
         tid += gridDim.x * blockDim.x) {
        int j    = tid & 15;
        int lane = (tid >> 4) & 31;
        int rest = tid >> 9;              // /(16*32)
        int kb   = rest % KB;
        int t    = rest / KB;
        int K = kb * 32 + ((lane < 16) ? 0 : 8) + ((j < 8) ? j : j + 8);
        int n = t * 16 + (lane & 15);
        out[tid] = f2bf(W[(size_t)K * NO + n]);
    }
}

// ---------------------------------------------------------------------------
// WMMA GEMM: OUT[rows,NO] = act(A[rows,NI](bf16) @ Wpacked + bias)
// One wave per 16x16 tile of C; blockDim = (NO/16)*32; gridDim.x = rows/16.
// A tile (16 x NI bf16, <= 4KB) is staged once per block in LDS; each of the
// (NO/16) waves reads its fragments via ds_load_b128 instead of re-fetching
// the same rows from global 8x. B (packed weights) streams from L2.
// ACT: 0 none, 1 relu, 2 sigmoid.  HASB/OUTF/OUTB are compile-time.
// ---------------------------------------------------------------------------
template <int NI, int NO, int ACT, bool HASB, bool OUTF, bool OUTB>
__global__ void wmma_gemm(const unsigned short* __restrict__ A,
                          const unsigned short* __restrict__ Wp,
                          const float* __restrict__ bias,
                          float* __restrict__ outF,
                          unsigned short* __restrict__ outB) {
    constexpr int KB = NI / 32;
    __shared__ unsigned short tileA[16 * NI];         // <= 4 KB

    const int lane  = threadIdx.x & 31;
    const int tileN = threadIdx.x >> 5;
    const int tileM = blockIdx.x;
    const int row16 = lane & 15;
    const int half  = lane >> 4;

    // cooperative A-tile stage: 2*NI uint4's, blockDim = 2*NO threads
    {
        const uint4* gsrc = (const uint4*)(A + (size_t)tileM * 16 * NI);
        uint4* lds = (uint4*)tileA;
        for (int i = threadIdx.x; i < 2 * NI; i += blockDim.x) lds[i] = gsrc[i];
    }
    __syncthreads();

    const unsigned short* arow  = tileA + row16 * NI;
    const unsigned short* wbase = Wp + ((size_t)tileN * KB) * (32 * 16);

    v8f c = {};
#pragma unroll
    for (int kb = 0; kb < KB; ++kb) {
        union { uint4 q[2]; v16bf v; } af;
        af.q[0] = *(const uint4*)(arow + kb * 32 + half * 8);        // K = half*8 .. +7
        af.q[1] = *(const uint4*)(arow + kb * 32 + 16 + half * 8);   // K = 16+half*8 ..
        union { uint4 q[2]; v16bf v; } bfr;
        const uint4* pb = (const uint4*)(wbase + ((size_t)kb * 32 + lane) * 16);
        bfr.q[0] = pb[0];
        bfr.q[1] = pb[1];
        c = __builtin_amdgcn_wmma_f32_16x16x32_bf16(
            /*neg_a=*/false, af.v, /*neg_b=*/false, bfr.v,
            /*c_mod=*/(short)0, c, /*reuse_a=*/false, /*reuse_b=*/false);
    }

    const int col = tileN * 16 + row16;
    const float bv = HASB ? bias[col] : 0.0f;
#pragma unroll
    for (int r = 0; r < 8; ++r) {
        int orow = tileM * 16 + half * 8 + r;
        float v = c[r] + bv;
        if (ACT == 1) v = v > 0.0f ? v : 0.0f;
        else if (ACT == 2) v = 1.0f / (1.0f + __expf(-v));
        size_t idx = (size_t)orow * NO + col;
        if (OUTF) outF[idx] = v;
        if (OUTB) outB[idx] = f2bf(v);
    }
}

// ---------------------------------------------------------------------------
// Per-node attention halves: alpha_s[i] = h[i,:]·a_src ; alpha_d[i] = h[i,:]·a_dst
// One wave per node.
// ---------------------------------------------------------------------------
template <int F>
__global__ void alpha_dots(const float* __restrict__ h,
                           const float* __restrict__ a_s, const float* __restrict__ a_d,
                           float* __restrict__ as_out, float* __restrict__ ad_out, int n) {
    int wid  = (int)((blockIdx.x * (size_t)blockDim.x + threadIdx.x) >> 5);
    int lane = threadIdx.x & 31;
    if (wid >= n) return;
    float s = 0.0f, d = 0.0f;
#pragma unroll
    for (int k = lane; k < F; k += WAVE) {
        float hv = h[(size_t)wid * F + k];
        s += hv * a_s[k];
        d += hv * a_d[k];
    }
#pragma unroll
    for (int off = 16; off > 0; off >>= 1) {
        s += __shfl_xor(s, off, WAVE);
        d += __shfl_xor(d, off, WAVE);
    }
    if (lane == 0) { as_out[wid] = s; ad_out[wid] = d; }
}

// ---------------------------------------------------------------------------
// Edge passes (E real edges + N self-loops; eid>=E => self loop eid-E)
// ---------------------------------------------------------------------------
__global__ void edge_score(const int* __restrict__ ei,
                           const float* __restrict__ as, const float* __restrict__ ad,
                           float* __restrict__ ebuf, unsigned* __restrict__ mmap,
                           int E, int E2) {
    int t = blockIdx.x * blockDim.x + threadIdx.x;
    if (t >= E2) return;
    int s, d;
    if (t < E) { s = ei[t]; d = ei[E + t]; } else { s = t - E; d = s; }
    float e = as[s] + ad[d];
    e = e > 0.0f ? e : 0.2f * e;              // leaky_relu, NEG_SLOPE=0.2
    ebuf[t] = e;
    atomicMax(&mmap[d], fmap(e));
}

__global__ void edge_exp(const int* __restrict__ ei, float* __restrict__ ebuf,
                         const unsigned* __restrict__ mmap, float* __restrict__ denom,
                         int E, int E2) {
    int t = blockIdx.x * blockDim.x + threadIdx.x;
    if (t >= E2) return;
    int d = (t < E) ? ei[E + t] : (t - E);
    float ex = __expf(ebuf[t] - funmap(mmap[d]));
    ebuf[t] = ex;
    atomicAdd(&denom[d], ex);
}

template <int F>
__global__ void edge_agg(const int* __restrict__ ei, const float* __restrict__ ebuf,
                         const float* __restrict__ denom, const float* __restrict__ h,
                         float* __restrict__ agg, int E, int E2) {
    size_t gid = blockIdx.x * (size_t)blockDim.x + threadIdx.x;
    int eid  = (int)(gid >> 5);
    int lane = (int)(gid & 31);
    if (eid >= E2) return;
    int s, d;
    if (eid < E) { s = ei[eid]; d = ei[E + eid]; } else { s = eid - E; d = s; }
    float w = ebuf[eid] / denom[d];
#pragma unroll
    for (int k = lane; k < F; k += WAVE)
        atomicAdd(&agg[(size_t)d * F + k], h[(size_t)s * F + k] * w);
}

// out = relu(agg + bias)  ->  bf16
__global__ void finalize_relu(const float* __restrict__ agg, const float* __restrict__ bias,
                              unsigned short* __restrict__ outb, int F, size_t total) {
    for (size_t i = blockIdx.x * (size_t)blockDim.x + threadIdx.x; i < total;
         i += (size_t)gridDim.x * blockDim.x) {
        float v = agg[i] + bias[i % F];
        v = v > 0.0f ? v : 0.0f;
        outb[i] = f2bf(v);
    }
}

// z = mu + eps * exp(0.5*logvar)  ->  bf16
__global__ void reparam(const float* __restrict__ mu, const float* __restrict__ lv,
                        const float* __restrict__ eps, unsigned short* __restrict__ zb,
                        size_t total) {
    for (size_t i = blockIdx.x * (size_t)blockDim.x + threadIdx.x; i < total;
         i += (size_t)gridDim.x * blockDim.x)
        zb[i] = f2bf(mu[i] + eps[i] * __expf(0.5f * lv[i]));
}

// ---------------------------------------------------------------------------
// Host launcher
// ---------------------------------------------------------------------------
extern "C" void kernel_launch(void* const* d_in, const int* in_sizes, int n_in,
                              void* d_out, int out_size, void* d_ws, size_t ws_size,
                              hipStream_t stream) {
    (void)in_sizes; (void)n_in; (void)out_size; (void)ws_size;
    const int N = 50000, E = 800000, E2 = E + N;

    const float* x     = (const float*)d_in[0];
    const int*   ei    = (const int*)  d_in[1];
    const float* eps   = (const float*)d_in[2];
    const float* W1    = (const float*)d_in[3];
    const float* as1   = (const float*)d_in[4];
    const float* ad1   = (const float*)d_in[5];
    const float* b1    = (const float*)d_in[6];
    const float* W2    = (const float*)d_in[7];
    const float* as2   = (const float*)d_in[8];
    const float* ad2   = (const float*)d_in[9];
    const float* b2    = (const float*)d_in[10];
    const float* W_mu  = (const float*)d_in[11];
    const float* b_mu  = (const float*)d_in[12];
    const float* W_lv  = (const float*)d_in[13];
    const float* b_lv  = (const float*)d_in[14];
    const float* W_d1  = (const float*)d_in[15];
    const float* b_d1  = (const float*)d_in[16];
    const float* W_d2  = (const float*)d_in[17];
    const float* b_d2  = (const float*)d_in[18];

    float* out   = (float*)d_out;
    float* recon = out;                       // [N,128]
    float* muO   = out + (size_t)N * 128;     // [N,64]
    float* lvO   = muO + (size_t)N * 64;      // [N,64]

    // ---- workspace carve-up (256B aligned) ----
    size_t off = 0;
    auto alloc = [&](size_t bytes) -> char* {
        char* p = (char*)d_ws + off;
        off += (bytes + 255) & ~(size_t)255;
        return p;
    };
    unsigned short* xb    = (unsigned short*)alloc((size_t)N * 128 * 2);
    unsigned short* W1p   = (unsigned short*)alloc(128 * 128 * 2);
    unsigned short* W2p   = (unsigned short*)alloc(128 * 64 * 2);
    unsigned short* Wmup  = (unsigned short*)alloc(64 * 64 * 2);
    unsigned short* Wlvp  = (unsigned short*)alloc(64 * 64 * 2);
    unsigned short* Wd1p  = (unsigned short*)alloc(64 * 128 * 2);
    unsigned short* Wd2p  = (unsigned short*)alloc(128 * 128 * 2);
    float*          h1    = (float*)alloc((size_t)N * 128 * 4);
    float*          h2    = (float*)alloc((size_t)N * 64 * 4);
    float*          agg1  = (float*)alloc((size_t)N * 128 * 4);
    float*          agg2  = (float*)alloc((size_t)N * 64 * 4);
    float*          alS   = (float*)alloc((size_t)N * 4);
    float*          alD   = (float*)alloc((size_t)N * 4);
    unsigned*       mmap  = (unsigned*)alloc((size_t)N * 4);
    float*          den   = (float*)alloc((size_t)N * 4);
    float*          ebuf  = (float*)alloc((size_t)E2 * 4);
    unsigned short* hb1   = (unsigned short*)alloc((size_t)N * 128 * 2);
    unsigned short* hb2   = (unsigned short*)alloc((size_t)N * 64 * 2);
    unsigned short* zb    = (unsigned short*)alloc((size_t)N * 64 * 2);
    unsigned short* db    = (unsigned short*)alloc((size_t)N * 128 * 2);

    const int GS = 4096;                 // grid-stride blocks
    const int rowBlocks = N / 16;        // 3125 (exact)
    const int eB  = (E2 + 255) / 256;
    const int eWB = (E2 * 32 + 255) / 256;
    const int nWB = (int)(((size_t)N * 32 + 255) / 256);

    // ---- prep: bf16 inputs + packed weights ----
    cvt_f32_bf16<<<GS, 256, 0, stream>>>(x, xb, (size_t)N * 128);
    pack_w<<<64, 256, 0, stream>>>(W1,  W1p,  128, 128);
    pack_w<<<64, 256, 0, stream>>>(W2,  W2p,  128, 64);
    pack_w<<<16, 256, 0, stream>>>(W_mu, Wmup, 64, 64);
    pack_w<<<16, 256, 0, stream>>>(W_lv, Wlvp, 64, 64);
    pack_w<<<32, 256, 0, stream>>>(W_d1, Wd1p, 64, 128);
    pack_w<<<64, 256, 0, stream>>>(W_d2, Wd2p, 128, 128);

    // ================= GAT layer 1 (128 -> 128) =================
    wmma_gemm<128, 128, 0, false, true, false>
        <<<rowBlocks, 256, 0, stream>>>(xb, W1p, nullptr, h1, nullptr);
    alpha_dots<128><<<nWB, 256, 0, stream>>>(h1, as1, ad1, alS, alD, N);
    zero_f32<<<GS, 256, 0, stream>>>((float*)mmap, N);
    zero_f32<<<GS, 256, 0, stream>>>(den, N);
    zero_f32<<<GS, 256, 0, stream>>>(agg1, (size_t)N * 128);
    edge_score<<<eB, 256, 0, stream>>>(ei, alS, alD, ebuf, mmap, E, E2);
    edge_exp<<<eB, 256, 0, stream>>>(ei, ebuf, mmap, den, E, E2);
    edge_agg<128><<<eWB, 256, 0, stream>>>(ei, ebuf, den, h1, agg1, E, E2);
    finalize_relu<<<GS, 256, 0, stream>>>(agg1, b1, hb1, 128, (size_t)N * 128);

    // ================= GAT layer 2 (128 -> 64) =================
    wmma_gemm<128, 64, 0, false, true, false>
        <<<rowBlocks, 128, 0, stream>>>(hb1, W2p, nullptr, h2, nullptr);
    alpha_dots<64><<<nWB, 256, 0, stream>>>(h2, as2, ad2, alS, alD, N);
    zero_f32<<<GS, 256, 0, stream>>>((float*)mmap, N);
    zero_f32<<<GS, 256, 0, stream>>>(den, N);
    zero_f32<<<GS, 256, 0, stream>>>(agg2, (size_t)N * 64);
    edge_score<<<eB, 256, 0, stream>>>(ei, alS, alD, ebuf, mmap, E, E2);
    edge_exp<<<eB, 256, 0, stream>>>(ei, ebuf, mmap, den, E, E2);
    edge_agg<64><<<eWB, 256, 0, stream>>>(ei, ebuf, den, h2, agg2, E, E2);
    finalize_relu<<<GS, 256, 0, stream>>>(agg2, b2, hb2, 64, (size_t)N * 64);

    // ================= VAE heads =================
    wmma_gemm<64, 64, 0, true, true, false>
        <<<rowBlocks, 128, 0, stream>>>(hb2, Wmup, b_mu, muO, nullptr);
    wmma_gemm<64, 64, 0, true, true, false>
        <<<rowBlocks, 128, 0, stream>>>(hb2, Wlvp, b_lv, lvO, nullptr);
    reparam<<<GS, 256, 0, stream>>>(muO, lvO, eps, zb, (size_t)N * 64);
    wmma_gemm<64, 128, 1, true, false, true>
        <<<rowBlocks, 256, 0, stream>>>(zb, Wd1p, b_d1, nullptr, db);
    wmma_gemm<128, 128, 2, true, true, false>
        <<<rowBlocks, 256, 0, stream>>>(db, Wd2p, b_d2, recon, nullptr);
}
